// FakeMatryoshkaLinear_16587163697891
// MI455X (gfx1250) — matrix operand order repaired and checked by reading, not compiled, not run on hardware
//
#include <hip/hip_runtime.h>
#include <hip/hip_bf16.h>

typedef __attribute__((ext_vector_type(16))) _Float16 v16h;
typedef __attribute__((ext_vector_type(8)))  _Float16 v8h;
typedef __attribute__((ext_vector_type(8)))  float    v8f;

#define BM 128
#define BN 128
#define BK 64
#define LDT 72   // padded LDS row stride in halves (144B: 16B-aligned, bank-conflict free)

// ---------------------------------------------------------------------------
// Pass 1: dequantize int8-coded weights (4-bit slice with round bit) and
// scatter through perm so the GEMM sees w'[o, perm[i]] = dequant(q[o,i]).
// This removes the input gather from the hot loop entirely.
// ---------------------------------------------------------------------------
__global__ __launch_bounds__(256) void dequant_permute_kernel(
    const int* __restrict__ qw, const float* __restrict__ scale,
    const float* __restrict__ zero, const int* __restrict__ perm,
    _Float16* __restrict__ w16, int OUT, int IN, int G)
{
    const int gs = IN / G;
    long long idx4   = (long long)blockIdx.x * blockDim.x + threadIdx.x;
    long long total4 = (long long)OUT * IN / 4;
    if (idx4 >= total4) return;
    long long base = idx4 * 4;
    int o = (int)(base / IN);
    int i = (int)(base % IN);          // gs % 4 == 0 -> all 4 in same group
    int g = i / gs;
    float s = scale[(long long)o * G + g];
    float z = zero [(long long)o * G + g];
    int4 q = *(const int4*)(qw + base);
    int4 p = *(const int4*)(perm + i);
    long long orow = (long long)o * IN;
    int qa[4] = {q.x, q.y, q.z, q.w};
    int pa[4] = {p.x, p.y, p.z, p.w};
#pragma unroll
    for (int j = 0; j < 4; ++j) {
        int fl = qa[j] >> 4;                 // drop = 8-4
        int rb = (qa[j] >> 3) & 1;           // round bit
        int sl = fl + rb; if (sl > 15) sl = 15;
        float w = s * ((float)(sl << 4) - z);
        w16[orow + pa[j]] = (_Float16)w;     // scatter: fold perm into weights
    }
}

// ---------------------------------------------------------------------------
// Pass 2: tiled WMMA GEMM  Y[M][N] = X[M][K] * W[N][K]^T + bias
// 256 threads = 8 wave32s, block tile 128x128, BK=64, wave tile 64x32
// (4x2 fragments of v_wmma_f32_16x16x32_f16).
// ---------------------------------------------------------------------------
__global__ __launch_bounds__(256) void wmma_gemm_bias_kernel(
    const float*    __restrict__ X,     // [M][K] fp32 activations
    const _Float16* __restrict__ W,     // [N][K] f16 dequantized+permuted weights
    const float*    __restrict__ bias,  // [N]
    float*          __restrict__ Y,     // [M][N]
    int M, int N, int K)
{
    __shared__ _Float16 As[BM * LDT];
    __shared__ _Float16 Bs[BN * LDT];

    const int t    = threadIdx.x;
    const int lane = t & 31;
    const int wave = t >> 5;
    const int wr   = wave >> 2;            // 0..1 : wave row (64 rows each)
    const int wc   = wave & 3;             // 0..3 : wave col (32 cols each)

    const long long m0 = (long long)blockIdx.y * BM;
    const long long n0 = (long long)blockIdx.x * BN;

    // global->LDS staging: 2 threads per tile row, 32 K-elements each
    const int lrow = t >> 1;               // 0..127
    const int lcol = (t & 1) * 32;         // 0 or 32
    const float*    xrow = X + (m0 + lrow) * (long long)K;
    const _Float16* wrow = W + (n0 + lrow) * (long long)K;

    float4 sa[8];                          // 32 fp32 of A (converted on store)
    uint4  sb[4];                          // 32 f16 of B

    auto loadA = [&](int k0) {
#pragma unroll
        for (int r = 0; r < 8; ++r)
            sa[r] = *(const float4*)(xrow + k0 + lcol + r * 4);
    };
    auto loadB = [&](int k0) {
#pragma unroll
        for (int r = 0; r < 4; ++r)
            sb[r] = *(const uint4*)(wrow + k0 + lcol + r * 8);
    };
    auto stageStore = [&]() {
#pragma unroll
        for (int r = 0; r < 4; ++r) {
            float4 f0 = sa[2 * r], f1 = sa[2 * r + 1];
            v8h h = { (_Float16)f0.x, (_Float16)f0.y, (_Float16)f0.z, (_Float16)f0.w,
                      (_Float16)f1.x, (_Float16)f1.y, (_Float16)f1.z, (_Float16)f1.w };
            *(v8h*)(&As[lrow * LDT + lcol + r * 8]) = h;
        }
#pragma unroll
        for (int r = 0; r < 4; ++r)
            *(uint4*)(&Bs[lrow * LDT + lcol + r * 8]) = sb[r];
    };

    const int l15  = lane & 15;
    const int hiHW = lane >> 4;            // 0 for lanes 0-15, 1 for lanes 16-31
    const int rowA = wr * 64;
    const int rowB = wc * 32;

    auto cat16 = [&](v8h lo, v8h hi) -> v16h {
        return __builtin_shufflevector(lo, hi,
            0,1,2,3,4,5,6,7,8,9,10,11,12,13,14,15);
    };
    // A fragment (16x32, ISA §7.12.2): lanes 0-15 row=l, K={0..7,16..23};
    //                                  lanes 16-31 row=l-16, K={8..15,24..31}
    auto loadFragA = [&](const _Float16* p) -> v16h {
        return cat16(*(const v8h*)(p), *(const v8h*)(p + 16));
    };
    // B fragment (32x16): lanes 0-15 col=l, K={0..7,16..23}; lanes 16-31 col=l-16, K={8..15,24..31}
    auto loadFragB = [&](const _Float16* p) -> v16h {
        return cat16(*(const v8h*)(p), *(const v8h*)(p + 16));
    };

    v8f acc[4][2] = {};

    loadA(0); loadB(0);
    for (int k0 = 0; k0 < K; k0 += BK) {
        __syncthreads();                   // previous compute done reading LDS
        stageStore();
        __syncthreads();                   // tiles visible
        if (k0 + BK < K) { loadA(k0 + BK); loadB(k0 + BK); }  // overlap VMEM w/ WMMA

#pragma unroll
        for (int ks = 0; ks < BK; ks += 32) {
            v16h a[4], b[2];
#pragma unroll
            for (int fi = 0; fi < 4; ++fi)
                a[fi] = loadFragA(&As[(rowA + fi * 16 + l15) * LDT + ks + hiHW * 8]);
#pragma unroll
            for (int fj = 0; fj < 2; ++fj)
                b[fj] = loadFragB(&Bs[(rowB + fj * 16 + l15) * LDT + ks + hiHW * 8]);
#pragma unroll
            for (int fi = 0; fi < 4; ++fi)
#pragma unroll
                for (int fj = 0; fj < 2; ++fj)
                    acc[fi][fj] = __builtin_amdgcn_wmma_f32_16x16x32_f16(
                        false, a[fi], false, b[fj],
                        (short)0, acc[fi][fj], false, false);
        }
    }

    // Epilogue: C/D layout — VGPR v: lanes 0-15 -> M=v, lanes 16-31 -> M=v+8
    const long long mBase = m0 + rowA;
    const long long nBase = n0 + rowB;
#pragma unroll
    for (int fi = 0; fi < 4; ++fi) {
#pragma unroll
        for (int fj = 0; fj < 2; ++fj) {
            long long col = nBase + fj * 16 + l15;
            float bv = bias[col];
            long long row = mBase + fi * 16 + hiHW * 8;
            float* yp = Y + row * (long long)N + col;
#pragma unroll
            for (int v = 0; v < 8; ++v)
                yp[(long long)v * N] = acc[fi][fj][v] + bv;
        }
    }
}

// ---------------------------------------------------------------------------
extern "C" void kernel_launch(void* const* d_in, const int* in_sizes, int n_in,
                              void* d_out, int out_size, void* d_ws, size_t ws_size,
                              hipStream_t stream) {
    const float* inp   = (const float*)d_in[0];   // [B,S,IN] fp32
    const int*   qw    = (const int*)  d_in[1];   // [OUT,IN] int32
    const float* scale = (const float*)d_in[2];   // [OUT,G]
    const float* zero  = (const float*)d_in[3];   // [OUT,G]
    const float* bias  = (const float*)d_in[4];   // [OUT]
    const int*   perm  = (const int*)  d_in[5];   // [IN]

    const int OUT_ = in_sizes[4];
    const int IN_  = in_sizes[5];
    const int G    = in_sizes[2] / OUT_;
    const int M    = in_sizes[0] / IN_;           // B*S

    _Float16* w16 = (_Float16*)d_ws;              // [OUT][IN] f16, 2*OUT*IN bytes

    long long total4 = (long long)OUT_ * IN_ / 4;
    int blocks1 = (int)((total4 + 255) / 256);
    dequant_permute_kernel<<<blocks1, 256, 0, stream>>>(
        qw, scale, zero, perm, w16, OUT_, IN_, G);

    dim3 grid(OUT_ / BN, M / BM);
    wmma_gemm_bias_kernel<<<grid, 256, 0, stream>>>(
        inp, w16, bias, (float*)d_out, M, OUT_, IN_);
}
